// EquiformerEnc_33036888441081
// MI455X (gfx1250) — matrix operand order repaired and checked by reading, not compiled, not run on hardware
//
#include <hip/hip_runtime.h>
#include <cstdint>
#include <cstddef>

typedef _Float16 half_t;
typedef _Float16 v8h  __attribute__((ext_vector_type(8)));
typedef _Float16 v16h __attribute__((ext_vector_type(16)));
typedef float    v8f  __attribute__((ext_vector_type(8)));

// ---------------- problem constants ----------------
constexpr int   kB = 8, kP = 2048, kN = kB * kP;
constexpr int   kK0 = 16, kK1 = 1024;
constexpr float kR0 = 0.06f, kR1 = 2.0f;
constexpr int   kNB0 = 64, kNB1 = 256;
constexpr float kEPS = 1e-8f;
constexpr int   kE0 = kN * kK0;     // 262144
constexpr int   kE1 = kB * kK1;     // 8192
constexpr int   kHeads = 4;
constexpr float kPI = 3.14159265358979323846f;
// layer 0 dims
constexpr int C0 = 64, H0 = 64, A0c = 8, V0c = 4, HV0 = 16, ECH0 = 32, HF0 = 64, COUT0 = 128;
// layer 1 dims
constexpr int C1 = 128, H1 = 128, A1c = 16, V1c = 8, HV1 = 32, ECH1 = 64, HF1 = 256, COUT1 = 128;
// chunking of layer-0 edge pipeline
constexpr int NCHUNK = 4;
constexpr int Nc = kN / NCHUNK;      // 4096 nodes
constexpr int Ec = Nc * kK0;         // 65536 edges

enum ParamIdx {
  RAD_W1 = 0, RAD_B1, RAD_W2, RAD_B2, RAD_W3, RAD_B3,
  ATT1_W0, ATT1_WR, ATT1_WI, ALPHA_W, ALPHA_VEC,
  ATT2_W0, ATT2_WR, ATT2_WI, PROJ_P0, PROJ_B0, PROJ_P1,
  NORM1_W, NORM2_W, FFN1_L0, FFN1_L1, FFN1_B,
  FFN2_L0, FFN2_L1, FFN2_B, SC_L0, SC_L1, SC_B
};

// ---------------- elementwise / setup kernels ----------------

__global__ void k_feat_init(const float* __restrict__ pcd, const float* __restrict__ w_in,
                            const float* __restrict__ b_in, float* __restrict__ coords,
                            float* __restrict__ x) {
  int t = blockIdx.x * blockDim.x + threadIdx.x;
  if (t >= kN * C0) return;
  int n = t / C0, c = t % C0;
  const float* p = pcd + (size_t)n * 6;
  if (c < 3) coords[(size_t)n * 3 + c] = p[c];
  float f = b_in[c] + p[3] * w_in[c] + p[4] * w_in[64 + c] + p[5] * w_in[128 + c];
  float* xp = x + (size_t)n * 4 * C0;
  xp[c] = f;
  xp[C0 + c] = 0.f;
  xp[2 * C0 + c] = 0.f;
  xp[3 * C0 + c] = 0.f;
}

__global__ void k_knn(const float* __restrict__ coords, int* __restrict__ nbr,
                      float* __restrict__ dist) {
  int n = blockIdx.x * blockDim.x + threadIdx.x;
  if (n >= kN) return;
  int base = (n / kP) * kP;
  float px = coords[(size_t)n * 3 + 0], py = coords[(size_t)n * 3 + 1], pz = coords[(size_t)n * 3 + 2];
  float bd[kK0]; int bi[kK0];
#pragma unroll
  for (int i = 0; i < kK0; ++i) { bd[i] = 3.4e38f; bi[i] = base; }
  for (int j = 0; j < kP; ++j) {
    int jn = base + j;
    if (jn == n) continue;
    float dx = coords[(size_t)jn * 3 + 0] - px;
    float dy = coords[(size_t)jn * 3 + 1] - py;
    float dz = coords[(size_t)jn * 3 + 2] - pz;
    float d2 = dx * dx + dy * dy + dz * dz;
    if (d2 < bd[kK0 - 1]) {
      bd[kK0 - 1] = d2; bi[kK0 - 1] = jn;
#pragma unroll
      for (int q = kK0 - 1; q > 0; --q) {
        if (bd[q] < bd[q - 1]) {
          float td = bd[q]; bd[q] = bd[q - 1]; bd[q - 1] = td;
          int   ti = bi[q]; bi[q] = bi[q - 1]; bi[q - 1] = ti;
        }
      }
    }
  }
#pragma unroll
  for (int k = 0; k < kK0; ++k) {
    nbr[(size_t)n * kK0 + k]  = bi[k];
    dist[(size_t)n * kK0 + k] = sqrtf(fmaxf(bd[k], 0.f));
  }
}

__global__ void k_sel1(const float* __restrict__ coords, int* __restrict__ nbr1,
                       float* __restrict__ dist1) {
  __shared__ float sd2[kP];
  int b = blockIdx.x;
  const float* cb = coords + (size_t)b * kP * 3;
  for (int i = threadIdx.x; i < kP; i += blockDim.x) {
    float x = cb[i * 3], y = cb[i * 3 + 1], z = cb[i * 3 + 2];
    sd2[i] = x * x + y * y + z * z;
  }
  __syncthreads();
  for (int i = threadIdx.x; i < kP; i += blockDim.x) {
    float di = sd2[i];
    int rank = 0;
    for (int j = 0; j < kP; ++j) {
      float dj = sd2[j];
      rank += (dj < di) || (dj == di && j < i);
    }
    if (rank < kK1) {
      nbr1[b * kK1 + rank]  = b * kP + i;
      dist1[b * kK1 + rank] = sqrtf(fmaxf(di, 0.f));
    }
  }
}

__global__ void k_wigner(const float* __restrict__ coords, const int* __restrict__ nbr,
                         int E, int K, int useDst, float* __restrict__ D) {
  int e = blockIdx.x * blockDim.x + threadIdx.x;
  if (e >= E) return;
  int s = nbr[e];
  float vx = coords[(size_t)s * 3 + 0], vy = coords[(size_t)s * 3 + 1], vz = coords[(size_t)s * 3 + 2];
  if (useDst) {
    int d = e / K;
    vx -= coords[(size_t)d * 3 + 0]; vy -= coords[(size_t)d * 3 + 1]; vz -= coords[(size_t)d * 3 + 2];
  }
  float inv = 1.f / (sqrtf(vx * vx + vy * vy + vz * vz) + kEPS);
  vx *= inv; vy *= inv; vz *= inv;
  float rx, ry, rz;
  if (fabsf(vz) < 0.99f) { rx = 0.f; ry = 0.f; rz = 1.f; }
  else                   { rx = 1.f; ry = 0.f; rz = 0.f; }
  float b1x = ry * vz - rz * vy, b1y = rz * vx - rx * vz, b1z = rx * vy - ry * vx;
  float binv = 1.f / (sqrtf(b1x * b1x + b1y * b1y + b1z * b1z) + kEPS);
  b1x *= binv; b1y *= binv; b1z *= binv;
  float b2x = vy * b1z - vz * b1y, b2y = vz * b1x - vx * b1z, b2z = vx * b1y - vy * b1x;
  float R[3][3] = { {b1x, b1y, b1z}, {vx, vy, vz}, {b2x, b2y, b2z} };
  const int perm[3] = {1, 2, 0};
  float* Dp = D + (size_t)e * 9;
#pragma unroll
  for (int i = 0; i < 3; ++i)
#pragma unroll
    for (int j = 0; j < 3; ++j)
      Dp[3 * i + j] = R[perm[i]][perm[j]];
}

// radial MLP (rbf fused), one block per edge
__global__ void k_rad(const float* __restrict__ dist, float cutoff, int nbasis, int ech,
                      int cout2,
                      const float* __restrict__ W1, const float* __restrict__ b1,
                      const float* __restrict__ W2, const float* __restrict__ b2,
                      const float* __restrict__ W3, const float* __restrict__ b3,
                      float* __restrict__ rad) {
  __shared__ float sg[256];
  __shared__ float t1[64];
  __shared__ float t2[64];
  int e = blockIdx.x;
  float d = dist[e];
  float env = 0.5f * (cosf(kPI * fminf(fmaxf(d / cutoff, 0.f), 1.f)) + 1.f);
  float width = cutoff / (float)nbasis;
  for (int i = threadIdx.x; i < nbasis; i += blockDim.x) {
    float cen = cutoff * (float)i / (float)(nbasis - 1);
    float z = (d - cen) / width;
    sg[i] = expf(-0.5f * z * z) * env;
  }
  __syncthreads();
  for (int j = threadIdx.x; j < ech; j += blockDim.x) {
    float acc = b1[j];
    for (int k = 0; k < nbasis; ++k) acc += sg[k] * W1[(size_t)k * ech + j];
    float s = 1.f / (1.f + expf(-acc));
    t1[j] = acc * s;
  }
  __syncthreads();
  for (int j = threadIdx.x; j < ech; j += blockDim.x) {
    float acc = b2[j];
    for (int k = 0; k < ech; ++k) acc += t1[k] * W2[(size_t)k * ech + j];
    float s = 1.f / (1.f + expf(-acc));
    t2[j] = acc * s;
  }
  __syncthreads();
  for (int j = threadIdx.x; j < cout2; j += blockDim.x) {
    float acc = b3[j];
    for (int k = 0; k < ech; ++k) acc += t2[k] * W3[(size_t)k * cout2 + j];
    rad[(size_t)e * cout2 + j] = acc;
  }
}

__global__ void k_rmsnorm(const float* __restrict__ x, const float* __restrict__ w,
                          float* __restrict__ xn, int Nn, int C) {
  int n = blockIdx.x * blockDim.x + threadIdx.x;
  if (n >= Nn) return;
  const float* xp = x + (size_t)n * 4 * C;
  float s0 = 0.f, s1 = 0.f;
  for (int c = 0; c < C; ++c) s0 += xp[c] * xp[c];
  for (int c = 0; c < 3 * C; ++c) { float v = xp[C + c]; s1 += v * v; }
  float n0 = sqrtf(s0 / (float)C + kEPS);
  float n1 = sqrtf(s1 / (float)(3 * C) + kEPS);
  float* op = xn + (size_t)n * 4 * C;
  for (int c = 0; c < C; ++c) op[c] = xp[c] / n0 * w[c];
  for (int m = 1; m < 4; ++m)
    for (int c = 0; c < C; ++c) op[m * C + c] = xp[m * C + c] / n1 * w[C + c];
}

// fused gather + rotation + radial scaling -> f16 GEMM-ready A matrices
__global__ void k_msg(const float* __restrict__ xnS, const float* __restrict__ xnD,
                      const int* __restrict__ nbr, const float* __restrict__ D,
                      const float* __restrict__ rad,
                      half_t* __restrict__ A0h, half_t* __restrict__ A1h,
                      int EcN, int e0, int K, int C) {
  int t = blockIdx.x * blockDim.x + threadIdx.x;
  if (t >= EcN * C) return;
  int e = t / C, c = t % C;
  int ge = e0 + e;
  int src = nbr[e];
  int dst = ge / K;
  const float* xs = xnS + (size_t)src * 4 * C;
  const float* xd = xnD + (size_t)dst * 4 * C;
  const float* Dm = D + (size_t)e * 9;
  float s0 = xs[c], s1 = xs[C + c], s2 = xs[2 * C + c], s3 = xs[3 * C + c];
  float d0 = xd[c], d1 = xd[C + c], d2v = xd[2 * C + c], d3 = xd[3 * C + c];
  float D00 = Dm[0], D01 = Dm[1], D02 = Dm[2];
  float D10 = Dm[3], D11 = Dm[4], D12 = Dm[5];
  float D20 = Dm[6], D21 = Dm[7], D22 = Dm[8];
  float sr1 = D00 * s1 + D01 * s2 + D02 * s3;
  float sr2 = D10 * s1 + D11 * s2 + D12 * s3;
  float sr3 = D20 * s1 + D21 * s2 + D22 * s3;
  float dr1 = D00 * d1 + D01 * d2v + D02 * d3;
  float dr2 = D10 * d1 + D11 * d2v + D12 * d3;
  float dr3 = D20 * d1 + D21 * d2v + D22 * d3;
  float rs = rad[(size_t)e * 2 * C + c];
  float rd = rad[(size_t)e * 2 * C + C + c];
  half_t* a0 = A0h + (size_t)e * 4 * C;        // [l0_s, l0_d, l2_s, l2_d]
  a0[c]         = (half_t)(s0 * rs);
  a0[C + c]     = (half_t)(d0 * rd);
  a0[2 * C + c] = (half_t)(sr2 * rs);
  a0[3 * C + c] = (half_t)(dr2 * rd);
  half_t* a1 = A1h + (size_t)(2 * e) * 2 * C;  // row 2e = l3, row 2e+1 = l1
  a1[c]         = (half_t)(sr3 * rs);
  a1[C + c]     = (half_t)(dr3 * rd);
  a1[2 * C + c] = (half_t)(sr1 * rs);
  a1[3 * C + c] = (half_t)(dr1 * rd);
}

// weight conversion: W (K,N) f32 -> Bt (N,K) f16
__global__ void k_wt(const float* __restrict__ W, half_t* __restrict__ Bt, int K, int Nn) {
  int t = blockIdx.x * blockDim.x + threadIdx.x;
  if (t >= K * Nn) return;
  int n = t / K, k = t % K;
  Bt[(size_t)n * K + k] = (half_t)W[(size_t)k * Nn + n];
}

// combined [Wr | Wi] (K,2H) -> Bt (2H,K) f16
__global__ void k_wt_ri(const float* __restrict__ Wr, const float* __restrict__ Wi,
                        half_t* __restrict__ Bt, int K, int H) {
  int t = blockIdx.x * blockDim.x + threadIdx.x;
  if (t >= K * 2 * H) return;
  int n = t / K, k = t % K;
  float v = (n < H) ? Wr[(size_t)k * H + n] : Wi[(size_t)k * H + (n - H)];
  Bt[(size_t)n * K + k] = (half_t)v;
}

// ---------------- WMMA GEMM: C(M,N) f32 = A(M,K) f16 * Bt(N,K)^T f16 ----------------
// M % 16 == 0, N % (16*NT) == 0, K % 32 == 0 guaranteed by construction.
// Register-blocked along N: one A fragment feeds NT WMMA accumulators.
template <int NT>
__global__ void k_gemm_nt(const half_t* __restrict__ A, const half_t* __restrict__ Bt,
                          float* __restrict__ C, int M, int K, int N) {
  int gw = (blockIdx.x * blockDim.x + threadIdx.x) >> 5;
  int lane = threadIdx.x & 31;
  int tilesN = N >> 4;
  int groupsN = tilesN / NT;
  int groups = (M >> 4) * groupsN;
  if (gw >= groups) return;
  int tm = gw / groupsN, tg = gw % groupsN;
  int r  = lane & 15;
  int hi = lane >> 4;
  const half_t* Ap = A + (size_t)(tm * 16 + r) * K + hi * 16;                // A frag: 16 contiguous K
  const half_t* Bp = Bt + (size_t)(tg * NT * 16 + r) * K + hi * 16;          // B frag: 16 contiguous K
  v8f zero = {};
  v8f acc[NT];
#pragma unroll
  for (int q = 0; q < NT; ++q) acc[q] = zero;
  for (int k0 = 0; k0 < K; k0 += 32) {
    v8h  alo = *(const v8h*)(Ap + k0);
    v8h  ahi = *(const v8h*)(Ap + k0 + 8);
    v16h av  = __builtin_shufflevector(alo, ahi, 0, 1, 2, 3, 4, 5, 6, 7,
                                       8, 9, 10, 11, 12, 13, 14, 15);
#pragma unroll
    for (int q = 0; q < NT; ++q) {
      v16h bv = *(const v16h*)(Bp + (size_t)q * 16 * K + k0);
      acc[q] = __builtin_amdgcn_wmma_f32_16x16x32_f16(false, av, false, bv,
                                                      (short)0, acc[q], false, false);
    }
  }
  int orow = tm * 16 + hi * 8;
#pragma unroll
  for (int q = 0; q < NT; ++q) {
    float* Cp = C + (size_t)orow * N + (tg * NT + q) * 16 + r;
#pragma unroll
    for (int i = 0; i < 8; ++i) Cp[(size_t)i * N] = acc[q][i];
  }
}

// epilogue: combine SO2 GEMM outputs, apply separable S2 gate, repack f16 for next GEMMs
__global__ void k_hepi(const float* __restrict__ G0, const float* __restrict__ G1,
                       half_t* __restrict__ hA0, half_t* __restrict__ hA1,
                       float* __restrict__ h0f, int EcN, int H) {
  int t = blockIdx.x * blockDim.x + threadIdx.x;
  if (t >= EcN * H) return;
  int e = t / H, c = t % H;
  float y0 = G0[(size_t)e * 2 * H + c];
  float y2 = G0[(size_t)e * 2 * H + H + c];
  const float* g1a = G1 + (size_t)(2 * e) * 2 * H;
  const float* g1b = G1 + (size_t)(2 * e + 1) * 2 * H;
  float yr = g1a[c] - g1b[H + c];
  float yi = g1a[H + c] + g1b[c];
  float s = 1.f / (1.f + expf(-y0));
  float h0 = y0 * s, h1 = yi * s, h2 = y2 * s, h3 = yr * s;
  hA0[(size_t)e * 2 * H + c]       = (half_t)h0;
  hA0[(size_t)e * 2 * H + H + c]   = (half_t)h2;
  hA1[(size_t)(2 * e) * H + c]     = (half_t)h3;
  hA1[(size_t)(2 * e + 1) * H + c] = (half_t)h1;
  h0f[(size_t)e * H + c] = h0;
}

__global__ void k_vepi(const float* __restrict__ Vb0, const float* __restrict__ Vb1,
                       float* __restrict__ v, int EcN, int HV) {
  int t = blockIdx.x * blockDim.x + threadIdx.x;
  if (t >= EcN * HV) return;
  int e = t / HV, c = t % HV;
  float v0 = Vb0[(size_t)e * 2 * HV + c];
  float v2 = Vb0[(size_t)e * 2 * HV + HV + c];
  const float* a = Vb1 + (size_t)(2 * e) * 2 * HV;
  const float* b = Vb1 + (size_t)(2 * e + 1) * 2 * HV;
  float vr = a[c] - b[HV + c];
  float vi = a[HV + c] + b[c];
  float* o = v + (size_t)e * 4 * HV;
  o[c] = v0; o[HV + c] = vi; o[2 * HV + c] = v2; o[3 * HV + c] = vr;
}

// alpha logits: blockDim = heads*A, one block per edge
__global__ void k_alpha(const float* __restrict__ h0f, const float* __restrict__ alphaW,
                        const float* __restrict__ alphaVec, float* __restrict__ logits,
                        int H, int A) {
  __shared__ float red[64];
  int e = blockIdx.x;
  int t = threadIdx.x;
  int HA = blockDim.x;
  const float* hp = h0f + (size_t)e * H;
  float acc = 0.f;
  for (int k = 0; k < H; ++k) acc += hp[k] * alphaW[(size_t)k * HA + t];
  acc = (acc > 0.f) ? acc : 0.2f * acc;
  int hh = t / A, aa = t % A;
  red[t] = acc * alphaVec[hh * A + aa];
  __syncthreads();
  if (aa == 0) {
    float s = 0.f;
    for (int q = 0; q < A; ++q) s += red[hh * A + q];
    logits[(size_t)e * kHeads + hh] = s;
  }
}

// softmax over K with radius mask; one block per (node, head)
__global__ void k_softmax(const float* __restrict__ logits, const float* __restrict__ dist,
                          float R, float* __restrict__ attn, int K) {
  __shared__ float red[128];
  int n = blockIdx.x / kHeads, h = blockIdx.x % kHeads;
  int tid = threadIdx.x;
  float mx = -3.4e38f;
  for (int k = tid; k < K; k += blockDim.x) {
    float m = (dist[(size_t)n * K + k] < R) ? 1.f : 0.f;
    float l = (m != 0.f) ? logits[((size_t)n * K + k) * kHeads + h] : -1.0e9f;
    mx = fmaxf(mx, l);
  }
  red[tid] = mx; __syncthreads();
  for (int s = blockDim.x / 2; s > 0; s >>= 1) {
    if (tid < s) red[tid] = fmaxf(red[tid], red[tid + s]);
    __syncthreads();
  }
  mx = red[0]; __syncthreads();
  float sum = 0.f;
  for (int k = tid; k < K; k += blockDim.x) {
    float m = (dist[(size_t)n * K + k] < R) ? 1.f : 0.f;
    float l = (m != 0.f) ? logits[((size_t)n * K + k) * kHeads + h] : -1.0e9f;
    sum += expf(l - mx);
  }
  red[tid] = sum; __syncthreads();
  for (int s = blockDim.x / 2; s > 0; s >>= 1) {
    if (tid < s) red[tid] += red[tid + s];
    __syncthreads();
  }
  float inv = 1.f / red[0];
  for (int k = tid; k < K; k += blockDim.x) {
    float m = (dist[(size_t)n * K + k] < R) ? 1.f : 0.f;
    float l = (m != 0.f) ? logits[((size_t)n * K + k) * kHeads + h] : -1.0e9f;
    attn[((size_t)n * K + k) * kHeads + h] = expf(l - mx) * inv * m;
  }
}

// attention-weighted aggregation fused with inverse (transpose) rotation
__global__ void k_agg(const float* __restrict__ attn, const float* __restrict__ v,
                      const float* __restrict__ D, float* __restrict__ agg,
                      int Nn, int K, int HV, int V) {
  int t = blockIdx.x * blockDim.x + threadIdx.x;
  if (t >= Nn * HV) return;
  int n = t / HV, c = t % HV;
  int h = c / V;
  float a0 = 0.f, a1 = 0.f, a2 = 0.f, a3 = 0.f;
  for (int k = 0; k < K; ++k) {
    int e = n * K + k;
    float w = attn[(size_t)e * kHeads + h];
    const float* vp = v + (size_t)e * 4 * HV;
    const float* Dm = D + (size_t)e * 9;
    float v1 = vp[HV + c], v2 = vp[2 * HV + c], v3 = vp[3 * HV + c];
    a0 += w * vp[c];
    a1 += w * (Dm[0] * v1 + Dm[3] * v2 + Dm[6] * v3);
    a2 += w * (Dm[1] * v1 + Dm[4] * v2 + Dm[7] * v3);
    a3 += w * (Dm[2] * v1 + Dm[5] * v2 + Dm[8] * v3);
  }
  float* ag = agg + (size_t)n * 4 * HV;
  ag[c] = a0; ag[HV + c] = a1; ag[2 * HV + c] = a2; ag[3 * HV + c] = a3;
}

__global__ void k_proj(const float* __restrict__ agg, const float* __restrict__ Pr0,
                       const float* __restrict__ bb0, const float* __restrict__ Pr1,
                       const float* __restrict__ xdst, float* __restrict__ y,
                       int Nn, int HV, int C) {
  int t = blockIdx.x * blockDim.x + threadIdx.x;
  if (t >= Nn * 4 * C) return;
  int n = t / (4 * C), rem = t % (4 * C), m = rem / C, c = rem % C;
  const float* ag = agg + (size_t)n * 4 * HV + m * HV;
  const float* W = (m == 0) ? Pr0 : Pr1;
  float acc = (m == 0) ? bb0[c] : 0.f;
  for (int k = 0; k < HV; ++k) acc += ag[k] * W[(size_t)k * C + c];
  if (xdst) acc += xdst[(size_t)n * 4 * C + m * C + c];
  y[(size_t)n * 4 * C + m * C + c] = acc;
}

__global__ void k_so3(const float* __restrict__ X, const float* __restrict__ Wl0,
                      const float* __restrict__ Wl1, const float* __restrict__ bias,
                      float* __restrict__ Y, int Nn, int Cin, int Cout) {
  int t = blockIdx.x * blockDim.x + threadIdx.x;
  if (t >= Nn * 4 * Cout) return;
  int n = t / (4 * Cout), rem = t % (4 * Cout), m = rem / Cout, j = rem % Cout;
  const float* xp = X + (size_t)n * 4 * Cin + m * Cin;
  const float* W = (m == 0) ? Wl0 : Wl1;
  float acc = (m == 0) ? bias[j] : 0.f;
  for (int k = 0; k < Cin; ++k) acc += xp[k] * W[(size_t)k * Cout + j];
  Y[t] = acc;
}

__global__ void k_act(float* __restrict__ X, int Nn, int C) {
  int t = blockIdx.x * blockDim.x + threadIdx.x;
  if (t >= Nn * C) return;
  int n = t / C, c = t % C;
  float* xp = X + (size_t)n * 4 * C;
  float l0 = xp[c];
  float s = 1.f / (1.f + expf(-l0));
  xp[c] = l0 * s;
  xp[C + c] *= s; xp[2 * C + c] *= s; xp[3 * C + c] *= s;
}

__global__ void k_add(const float* __restrict__ a, const float* __restrict__ b,
                      float* __restrict__ o, int n) {
  int t = blockIdx.x * blockDim.x + threadIdx.x;
  if (t < n) o[t] = a[t] + b[t];
}

__global__ void k_fill0(float* __restrict__ p, int n) {
  int t = blockIdx.x * blockDim.x + threadIdx.x;
  if (t < n) p[t] = 0.f;
}

__global__ void k_out(const float* __restrict__ out2, float* __restrict__ dout) {
  int t = blockIdx.x * blockDim.x + threadIdx.x;
  if (t >= kB * COUT1 * 3) return;
  int b = t / (COUT1 * 3), rem = t % (COUT1 * 3), c = rem / 3, i = rem % 3;
  dout[t] = out2[(size_t)b * 4 * COUT1 + (1 + i) * COUT1 + c];
}

// ---------------- host orchestration ----------------
static inline int cdiv(long long a, long long b) { return (int)((a + b - 1) / b); }

extern "C" void kernel_launch(void* const* d_in, const int* in_sizes, int n_in,
                              void* d_out, int out_size, void* d_ws, size_t ws_size,
                              hipStream_t stream) {
  (void)in_sizes; (void)n_in; (void)out_size; (void)ws_size;
  const float* pcd  = (const float*)d_in[0];
  const float* w_in = (const float*)d_in[1];
  const float* b_in = (const float*)d_in[2];
  auto p0 = [&](int i) -> const float* { return (const float*)d_in[3 + i]; };
  auto p1 = [&](int i) -> const float* { return (const float*)d_in[31 + i]; };

  char* wsb = (char*)d_ws;
  size_t cur = 0;
  auto alloc = [&](size_t bytes) -> void* {
    void* p = wsb + cur;
    cur += (bytes + 255) & ~(size_t)255;
    return p;
  };

  // persistent
  float*  coords = (float*)alloc((size_t)kN * 3 * 4);
  float*  x0buf  = (float*)alloc((size_t)kN * 4 * C0 * 4);
  int*    nbr0   = (int*)alloc((size_t)kE0 * 4);
  float*  dist0  = (float*)alloc((size_t)kE0 * 4);
  float*  D0     = (float*)alloc((size_t)kE0 * 9 * 4);
  float*  xn0    = (float*)alloc((size_t)kN * 4 * C0 * 4);
  float*  agg0   = (float*)alloc((size_t)kN * 4 * HV0 * 4);
  float*  y0buf  = (float*)alloc((size_t)kN * 4 * C0 * 4);
  float*  yn0    = (float*)alloc((size_t)kN * 4 * C0 * 4);
  float*  hf0    = (float*)alloc((size_t)kN * 4 * HF0 * 4);
  float*  ffn0   = (float*)alloc((size_t)kN * 4 * COUT0 * 4);
  float*  sc0    = (float*)alloc((size_t)kN * 4 * COUT0 * 4);
  float*  x1buf  = (float*)alloc((size_t)kN * 4 * COUT0 * 4);
  // f16 transposed weights
  half_t* W0t1   = (half_t*)alloc((size_t)(2 * H0) * (4 * C0) * 2);
  half_t* WRIt1  = (half_t*)alloc((size_t)(2 * H0) * (2 * C0) * 2);
  half_t* W0t2   = (half_t*)alloc((size_t)(2 * HV0) * (2 * H0) * 2);
  half_t* WRIt2  = (half_t*)alloc((size_t)(2 * HV0) * (H0) * 2);
  half_t* W0t1b  = (half_t*)alloc((size_t)(2 * H1) * (4 * C1) * 2);
  half_t* WRIt1b = (half_t*)alloc((size_t)(2 * H1) * (2 * C1) * 2);
  half_t* W0t2b  = (half_t*)alloc((size_t)(2 * HV1) * (2 * H1) * 2);
  half_t* WRIt2b = (half_t*)alloc((size_t)(2 * HV1) * (H1) * 2);
  // chunk scratch (layer0 sizes)
  float*  rad_c    = (float*)alloc((size_t)Ec * 2 * C0 * 4);
  half_t* A0h_c    = (half_t*)alloc((size_t)Ec * 4 * C0 * 2);
  half_t* A1h_c    = (half_t*)alloc((size_t)2 * Ec * 2 * C0 * 2);
  float*  G0_c     = (float*)alloc((size_t)Ec * 2 * H0 * 4);
  float*  G1_c     = (float*)alloc((size_t)2 * Ec * 2 * H0 * 4);
  half_t* hA0_c    = (half_t*)alloc((size_t)Ec * 2 * H0 * 2);
  half_t* hA1_c    = (half_t*)alloc((size_t)2 * Ec * H0 * 2);
  float*  h0f_c    = (float*)alloc((size_t)Ec * H0 * 4);
  float*  V0_c     = (float*)alloc((size_t)Ec * 2 * HV0 * 4);
  float*  V1_c     = (float*)alloc((size_t)2 * Ec * 2 * HV0 * 4);
  float*  v_c      = (float*)alloc((size_t)Ec * 4 * HV0 * 4);
  float*  logits_c = (float*)alloc((size_t)Ec * kHeads * 4);
  float*  attn_c   = (float*)alloc((size_t)Ec * kHeads * 4);
  // layer 1
  int*    nbr1    = (int*)alloc((size_t)kE1 * 4);
  float*  dist1   = (float*)alloc((size_t)kE1 * 4);
  float*  D1      = (float*)alloc((size_t)kE1 * 9 * 4);
  float*  xn1     = (float*)alloc((size_t)kN * 4 * C1 * 4);
  float*  xn1d    = (float*)alloc((size_t)kB * 4 * C1 * 4);
  float*  rad1    = (float*)alloc((size_t)kE1 * 2 * C1 * 4);
  half_t* A0h1    = (half_t*)alloc((size_t)kE1 * 4 * C1 * 2);
  half_t* A1h1    = (half_t*)alloc((size_t)2 * kE1 * 2 * C1 * 2);
  float*  G0b     = (float*)alloc((size_t)kE1 * 2 * H1 * 4);
  float*  G1b     = (float*)alloc((size_t)2 * kE1 * 2 * H1 * 4);
  half_t* hA0b    = (half_t*)alloc((size_t)kE1 * 2 * H1 * 2);
  half_t* hA1b    = (half_t*)alloc((size_t)2 * kE1 * H1 * 2);
  float*  h0f1    = (float*)alloc((size_t)kE1 * H1 * 4);
  float*  V0b1    = (float*)alloc((size_t)kE1 * 2 * HV1 * 4);
  float*  V1b1    = (float*)alloc((size_t)2 * kE1 * 2 * HV1 * 4);
  float*  v1      = (float*)alloc((size_t)kE1 * 4 * HV1 * 4);
  float*  logits1 = (float*)alloc((size_t)kE1 * kHeads * 4);
  float*  attn1   = (float*)alloc((size_t)kE1 * kHeads * 4);
  float*  agg1    = (float*)alloc((size_t)kB * 4 * HV1 * 4);
  float*  y1      = (float*)alloc((size_t)kB * 4 * C1 * 4);
  float*  yn1     = (float*)alloc((size_t)kB * 4 * C1 * 4);
  float*  hf1     = (float*)alloc((size_t)kB * 4 * HF1 * 4);
  float*  ffn1o   = (float*)alloc((size_t)kB * 4 * COUT1 * 4);
  float*  sc1     = (float*)alloc((size_t)kB * 4 * COUT1 * 4);
  float*  out2    = (float*)alloc((size_t)kB * 4 * COUT1 * 4);

  auto gemm = [&](const half_t* A, const half_t* Bt, float* Cc, int M, int K, int N) {
    int tilesN = N / 16;
    if (tilesN % 4 == 0) {
      long long groups = (long long)(M / 16) * (tilesN / 4);
      k_gemm_nt<4><<<cdiv(groups * 32, 256), 256, 0, stream>>>(A, Bt, Cc, M, K, N);
    } else {
      long long groups = (long long)(M / 16) * (tilesN / 2);
      k_gemm_nt<2><<<cdiv(groups * 32, 256), 256, 0, stream>>>(A, Bt, Cc, M, K, N);
    }
  };
  auto wt = [&](const float* W, half_t* Bt, int K, int N) {
    k_wt<<<cdiv((long long)K * N, 256), 256, 0, stream>>>(W, Bt, K, N);
  };
  auto wtri = [&](const float* Wr, const float* Wi, half_t* Bt, int K, int H) {
    k_wt_ri<<<cdiv((long long)K * 2 * H, 256), 256, 0, stream>>>(Wr, Wi, Bt, K, H);
  };

  // weight prep
  wt(p0(ATT1_W0), W0t1, 4 * C0, 2 * H0);
  wtri(p0(ATT1_WR), p0(ATT1_WI), WRIt1, 2 * C0, H0);
  wt(p0(ATT2_W0), W0t2, 2 * H0, 2 * HV0);
  wtri(p0(ATT2_WR), p0(ATT2_WI), WRIt2, H0, HV0);
  wt(p1(ATT1_W0), W0t1b, 4 * C1, 2 * H1);
  wtri(p1(ATT1_WR), p1(ATT1_WI), WRIt1b, 2 * C1, H1);
  wt(p1(ATT2_W0), W0t2b, 2 * H1, 2 * HV1);
  wtri(p1(ATT2_WR), p1(ATT2_WI), WRIt2b, H1, HV1);

  // ---- setup + geometry ----
  k_feat_init<<<cdiv((long long)kN * C0, 256), 256, 0, stream>>>(pcd, w_in, b_in, coords, x0buf);
  k_knn<<<cdiv(kN, 128), 128, 0, stream>>>(coords, nbr0, dist0);
  k_wigner<<<cdiv(kE0, 256), 256, 0, stream>>>(coords, nbr0, kE0, kK0, 1, D0);
  k_rmsnorm<<<cdiv(kN, 128), 128, 0, stream>>>(x0buf, p0(NORM1_W), xn0, kN, C0);

  // ---- layer 0, chunked edge pipeline ----
  for (int ci = 0; ci < NCHUNK; ++ci) {
    int n0 = ci * Nc;
    int e0 = n0 * kK0;
    const float* distC = dist0 + e0;
    const int*   nbrC  = nbr0 + e0;
    const float* DC    = D0 + (size_t)e0 * 9;
    k_rad<<<Ec, 128, 0, stream>>>(distC, kR0 * 0.99f, kNB0, ECH0, 2 * C0,
                                  p0(RAD_W1), p0(RAD_B1), p0(RAD_W2), p0(RAD_B2),
                                  p0(RAD_W3), p0(RAD_B3), rad_c);
    k_msg<<<cdiv((long long)Ec * C0, 256), 256, 0, stream>>>(xn0, xn0, nbrC, DC, rad_c,
                                                             A0h_c, A1h_c, Ec, e0, kK0, C0);
    gemm(A0h_c, W0t1, G0_c, Ec, 4 * C0, 2 * H0);
    gemm(A1h_c, WRIt1, G1_c, 2 * Ec, 2 * C0, 2 * H0);
    k_hepi<<<cdiv((long long)Ec * H0, 256), 256, 0, stream>>>(G0_c, G1_c, hA0_c, hA1_c, h0f_c, Ec, H0);
    gemm(hA0_c, W0t2, V0_c, Ec, 2 * H0, 2 * HV0);
    gemm(hA1_c, WRIt2, V1_c, 2 * Ec, H0, 2 * HV0);
    k_vepi<<<cdiv((long long)Ec * HV0, 256), 256, 0, stream>>>(V0_c, V1_c, v_c, Ec, HV0);
    k_alpha<<<Ec, kHeads * A0c, 0, stream>>>(h0f_c, p0(ALPHA_W), p0(ALPHA_VEC), logits_c, H0, A0c);
    k_softmax<<<Nc * kHeads, 128, 0, stream>>>(logits_c, distC, kR0, attn_c, kK0);
    k_agg<<<cdiv((long long)Nc * HV0, 256), 256, 0, stream>>>(attn_c, v_c, DC,
                                                              agg0 + (size_t)n0 * 4 * HV0,
                                                              Nc, kK0, HV0, V0c);
  }

  // ---- layer 0, per-node tail ----
  k_proj<<<cdiv((long long)kN * 4 * C0, 256), 256, 0, stream>>>(agg0, p0(PROJ_P0), p0(PROJ_B0),
                                                                p0(PROJ_P1), x0buf, y0buf, kN, HV0, C0);
  k_rmsnorm<<<cdiv(kN, 128), 128, 0, stream>>>(y0buf, p0(NORM2_W), yn0, kN, C0);
  k_so3<<<cdiv((long long)kN * 4 * HF0, 256), 256, 0, stream>>>(yn0, p0(FFN1_L0), p0(FFN1_L1),
                                                                p0(FFN1_B), hf0, kN, C0, HF0);
  k_act<<<cdiv((long long)kN * HF0, 256), 256, 0, stream>>>(hf0, kN, HF0);
  k_so3<<<cdiv((long long)kN * 4 * COUT0, 256), 256, 0, stream>>>(hf0, p0(FFN2_L0), p0(FFN2_L1),
                                                                  p0(FFN2_B), ffn0, kN, HF0, COUT0);
  k_so3<<<cdiv((long long)kN * 4 * COUT0, 256), 256, 0, stream>>>(y0buf, p0(SC_L0), p0(SC_L1),
                                                                  p0(SC_B), sc0, kN, C0, COUT0);
  k_add<<<cdiv((long long)kN * 4 * COUT0, 256), 256, 0, stream>>>(sc0, ffn0, x1buf, kN * 4 * COUT0);

  // ---- layer 1 ----
  k_sel1<<<kB, 256, 0, stream>>>(coords, nbr1, dist1);
  k_wigner<<<cdiv(kE1, 256), 256, 0, stream>>>(coords, nbr1, kE1, kK1, 0, D1);
  k_rmsnorm<<<cdiv(kN, 128), 128, 0, stream>>>(x1buf, p1(NORM1_W), xn1, kN, C1);
  k_fill0<<<cdiv(kB * 4 * C1, 256), 256, 0, stream>>>(xn1d, kB * 4 * C1);
  k_rad<<<kE1, 128, 0, stream>>>(dist1, kR1 * 0.99f, kNB1, ECH1, 2 * C1,
                                 p1(RAD_W1), p1(RAD_B1), p1(RAD_W2), p1(RAD_B2),
                                 p1(RAD_W3), p1(RAD_B3), rad1);
  k_msg<<<cdiv((long long)kE1 * C1, 256), 256, 0, stream>>>(xn1, xn1d, nbr1, D1, rad1,
                                                            A0h1, A1h1, kE1, 0, kK1, C1);
  gemm(A0h1, W0t1b, G0b, kE1, 4 * C1, 2 * H1);
  gemm(A1h1, WRIt1b, G1b, 2 * kE1, 2 * C1, 2 * H1);
  k_hepi<<<cdiv((long long)kE1 * H1, 256), 256, 0, stream>>>(G0b, G1b, hA0b, hA1b, h0f1, kE1, H1);
  gemm(hA0b, W0t2b, V0b1, kE1, 2 * H1, 2 * HV1);
  gemm(hA1b, WRIt2b, V1b1, 2 * kE1, H1, 2 * HV1);
  k_vepi<<<cdiv((long long)kE1 * HV1, 256), 256, 0, stream>>>(V0b1, V1b1, v1, kE1, HV1);
  k_alpha<<<kE1, kHeads * A1c, 0, stream>>>(h0f1, p1(ALPHA_W), p1(ALPHA_VEC), logits1, H1, A1c);
  k_softmax<<<kB * kHeads, 128, 0, stream>>>(logits1, dist1, kR1, attn1, kK1);
  k_agg<<<cdiv((long long)kB * HV1, 256), 256, 0, stream>>>(attn1, v1, D1, agg1, kB, kK1, HV1, V1c);
  k_proj<<<cdiv((long long)kB * 4 * C1, 256), 256, 0, stream>>>(agg1, p1(PROJ_P0), p1(PROJ_B0),
                                                                p1(PROJ_P1), nullptr, y1, kB, HV1, C1);
  k_rmsnorm<<<1, 128, 0, stream>>>(y1, p1(NORM2_W), yn1, kB, C1);
  k_so3<<<cdiv((long long)kB * 4 * HF1, 256), 256, 0, stream>>>(yn1, p1(FFN1_L0), p1(FFN1_L1),
                                                                p1(FFN1_B), hf1, kB, C1, HF1);
  k_act<<<cdiv((long long)kB * HF1, 256), 256, 0, stream>>>(hf1, kB, HF1);
  k_so3<<<cdiv((long long)kB * 4 * COUT1, 256), 256, 0, stream>>>(hf1, p1(FFN2_L0), p1(FFN2_L1),
                                                                  p1(FFN2_B), ffn1o, kB, HF1, COUT1);
  k_so3<<<cdiv((long long)kB * 4 * COUT1, 256), 256, 0, stream>>>(y1, p1(SC_L0), p1(SC_L1),
                                                                  p1(SC_B), sc1, kB, C1, COUT1);
  k_add<<<cdiv((long long)kB * 4 * COUT1, 256), 256, 0, stream>>>(sc1, ffn1o, out2, kB * 4 * COUT1);
  k_out<<<cdiv((long long)kB * COUT1 * 3, 256), 256, 0, stream>>>(out2, (float*)d_out);
}
